// MultiHeadDotProductAttention_83434034692505
// MI455X (gfx1250) — matrix operand order repaired and checked by reading, not compiled, not run on hardware
//
#include <hip/hip_runtime.h>
#include <hip/hip_bf16.h>

typedef __attribute__((ext_vector_type(16))) _Float16 v16h;
typedef __attribute__((ext_vector_type(8)))  _Float16 v8h;
typedef __attribute__((ext_vector_type(8)))  float    v8f;

union V16 { v16h v; v8h h[2]; };

// CDNA5 async global->LDS DMA (ASYNCcnt-tracked) via inline asm:
// VDST = LDS byte address VGPR, VADDR = 64-bit global address, SADDR = off.
#define USE_ASYNC_LDS 1

__device__ __forceinline__ void cp16B(_Float16* lds_dst, const _Float16* g_src) {
#if USE_ASYNC_LDS
    // generic->LDS aperture mapping: LDS offset is the low 32 bits of the generic address
    unsigned lds_off = (unsigned)(unsigned long long)lds_dst;
    asm volatile("global_load_async_to_lds_b128 %0, %1, off"
                 :
                 : "v"(lds_off), "v"(g_src)
                 : "memory");
#else
    *(v8h*)lds_dst = *(const v8h*)g_src;
#endif
}
__device__ __forceinline__ void async_wait_all() {
#if USE_ASYNC_LDS
    asm volatile("s_wait_asynccnt 0x0" ::: "memory");
#endif
}

// ---------------------------------------------------------------------------
// f32 -> f16 conversion (grid-stride)
// ---------------------------------------------------------------------------
__global__ void cvt_f32_to_f16_kernel(const float* __restrict__ in,
                                      _Float16* __restrict__ out, int n) {
    int i = blockIdx.x * blockDim.x + threadIdx.x;
    int stride = gridDim.x * blockDim.x;
    for (; i < n; i += stride) out[i] = (_Float16)in[i];
}

// ---------------------------------------------------------------------------
// GEMM: C[M,N] = scale * (A[M,K] * B[K,N] + bias[N]);  A,B f16 row-major.
// Block tile 128x128, 256 threads (8 waves). Wave w: rows (w>>1)*32 (2 M-tiles),
// cols (w&1)*64 (4 N-tiles) -> 8 WMMAs per 32-deep k-step, A frags reused 4x.
// Double-buffered LDS: async DMA for step k+1 overlaps WMMAs of step k.
// ---------------------------------------------------------------------------
template <typename TOUT>
__global__ __launch_bounds__(256) void gemm_bias_kernel(
    const _Float16* __restrict__ A, const _Float16* __restrict__ B,
    const float* __restrict__ bias, TOUT* __restrict__ C,
    int M, int N, int K, float scale)
{
    __shared__ __align__(16) _Float16 As[2][128 * 32];
    __shared__ __align__(16) _Float16 Bs[2][32 * 128];

    const int tid  = threadIdx.x;
    const int wave = tid >> 5;
    const int lane = tid & 31;
    const int l16  = lane & 15;
    const int hi   = lane >> 4;
    const int mblk = (int)blockIdx.x * 128;
    const int nblk = (int)blockIdx.y * 128;
    const int wr   = (wave >> 1) * 32;   // wave M offset (2 tiles)
    const int wc   = (wave & 1) * 64;    // wave N offset (4 tiles)

    auto stage = [&](int kt, int buf) {
#pragma unroll
        for (int i = 0; i < 2; ++i) {           // A: 128x32 halves
            int c = tid + i * 256;
            int row = c >> 2, col = (c & 3) * 8;
            cp16B(&As[buf][row * 32 + col], &A[(size_t)(mblk + row) * K + kt + col]);
        }
#pragma unroll
        for (int i = 0; i < 2; ++i) {           // B: 32x128 halves
            int c = tid + i * 256;
            int row = c >> 4, col = (c & 15) * 8;
            cp16B(&Bs[buf][row * 128 + col], &B[(size_t)(kt + row) * N + nblk + col]);
        }
    };

    v8f acc[2][4];
#pragma unroll
    for (int mt = 0; mt < 2; ++mt)
#pragma unroll
        for (int nt = 0; nt < 4; ++nt) acc[mt][nt] = (v8f){};

    const int nk = K >> 5;
    stage(0, 0);                                 // prologue

    for (int ki = 0; ki < nk; ++ki) {
        async_wait_all();       // current buffer's DMA retired
        __syncthreads();        // ... and visible to all waves
        if (ki + 1 < nk) stage((ki + 1) * 32, (ki + 1) & 1);   // overlap next DMA

        const _Float16* Ab = As[ki & 1];
        const _Float16* Bb = Bs[ki & 1];

        // A fragments (lane m = l16; halves k = hi*8+e and 16+hi*8+e)
        V16 a[2];
#pragma unroll
        for (int mt = 0; mt < 2; ++mt) {
            const int arow = (wr + mt * 16 + l16) * 32;
            a[mt].h[0] = *(const v8h*)&Ab[arow + hi * 8];
            a[mt].h[1] = *(const v8h*)&Ab[arow + 16 + hi * 8];
        }
        // B fragments (lane n = l16; halves k = hi*8+e and 16+hi*8+e)
        const int ncol = wc + l16;
#pragma unroll
        for (int nt = 0; nt < 4; ++nt) {
            V16 b;
#pragma unroll
            for (int e = 0; e < 8; ++e) {
                b.v[e]     = Bb[(hi * 8 + e) * 128 + ncol + nt * 16];
                b.v[e + 8] = Bb[(16 + hi * 8 + e) * 128 + ncol + nt * 16];
            }
#pragma unroll
            for (int mt = 0; mt < 2; ++mt)
                acc[mt][nt] = __builtin_amdgcn_wmma_f32_16x16x32_f16(
                    false, a[mt].v, false, b.v, (short)0, acc[mt][nt], false, false);
        }
    }

    // epilogue: lane holds col n = l16 (per N tile), rows m = r + 8*hi
#pragma unroll
    for (int nt = 0; nt < 4; ++nt) {
        const int n = nblk + wc + nt * 16 + l16;
        const float bb = bias[n];
#pragma unroll
        for (int mt = 0; mt < 2; ++mt) {
#pragma unroll
            for (int r = 0; r < 8; ++r) {
                int m = mblk + wr + mt * 16 + r + hi * 8;
                C[(size_t)m * N + n] = (TOUT)((acc[mt][nt][r] + bb) * scale);
            }
        }
    }
}

// ---------------------------------------------------------------------------
// Flash attention, S-transposed formulation. One block per (b, h, 64-query
// tile), 128 threads = 4 waves; wave w owns 16 query COLUMNS w*16..w*16+15.
//   S^T[key][q] = K . Q^T   (K staged naturally [key][d]; Q^T staged once)
//   The S^T C-fragment layout == P A-fragment layout, so P stays in registers.
//   O[q][d] += P . V        (V staged naturally [key][d])
// Double-buffered K/V tiles: async DMA for tile t+1 overlaps tile t compute.
// Q pre-scaled by 1/8. Layouts: [B, L, H, 64] f16, token stride = H*64.
// ---------------------------------------------------------------------------
__global__ __launch_bounds__(128) void flash_attn_kernel(
    const _Float16* __restrict__ Q, const _Float16* __restrict__ Kg,
    const _Float16* __restrict__ V, _Float16* __restrict__ O,
    int L, int H)
{
    constexpr int Hd = 64;
    __shared__ __align__(16) _Float16 Qt[64 * 64];        // transposed: [d][q]
    __shared__ __align__(16) _Float16 Ks[2][64 * 64];     // natural:    [key][d]
    __shared__ __align__(16) _Float16 Vs[2][64 * 64];     // natural:    [key][d]

    const int tid  = threadIdx.x;
    const int wave = tid >> 5;
    const int lane = tid & 31;
    const int l16  = lane & 15;
    const int hi   = lane >> 4;

    const int nqt = L / 64;
    const int qt  = (int)blockIdx.x % nqt;
    const int h   = ((int)blockIdx.x / nqt) % H;
    const int b   = (int)blockIdx.x / (nqt * H);

    const size_t tokStride = (size_t)H * Hd;
    const size_t headBase  = ((size_t)b * L) * tokStride + (size_t)h * Hd;

    auto stageKV = [&](int ktile, int buf) {
#pragma unroll
        for (int i = 0; i < 4; ++i) {
            int c = tid + i * 128;
            int key = c >> 3, d0 = (c & 7) * 8;
            size_t gtok = headBase + (size_t)(ktile * 64 + key) * tokStride + d0;
            cp16B(&Ks[buf][key * 64 + d0], &Kg[gtok]);
            cp16B(&Vs[buf][key * 64 + d0], &V[gtok]);
        }
    };

    // stage Q^T once (scatter-transpose) and kick off DMA for key-tile 0
    for (int c = tid; c < 64 * 8; c += 128) {
        int qrow = c >> 3;
        int d0   = (c & 7) * 8;
        v8h qv = *(const v8h*)&Q[headBase + (size_t)(qt * 64 + qrow) * tokStride + d0];
#pragma unroll
        for (int j = 0; j < 8; ++j) Qt[(d0 + j) * 64 + qrow] = qv[j];
    }
    stageKV(0, 0);
    __syncthreads();    // Qt visible

    // hoist loop-invariant Q^T B-fragments (lane q col, d = hi*8+e and 16+hi*8+e)
    V16 bq[2];
#pragma unroll
    for (int dt = 0; dt < 2; ++dt) {
        const int qcol = wave * 16 + l16;
#pragma unroll
        for (int e = 0; e < 8; ++e) {
            bq[dt].v[e]     = Qt[(dt * 32 + hi * 8 + e) * 64 + qcol];
            bq[dt].v[e + 8] = Qt[(dt * 32 + 16 + hi * 8 + e) * 64 + qcol];
        }
    }

    v8f o[4];
#pragma unroll
    for (int nt = 0; nt < 4; ++nt) o[nt] = (v8f){};
    float rmax = -__builtin_inff();
    float rsum = 0.f;

    const int ntiles = L / 64;
    for (int ktile = 0; ktile < ntiles; ++ktile) {
        async_wait_all();      // current tile's DMA retired
        __syncthreads();       // ... and visible; other waves done with old buf
        if (ktile + 1 < ntiles) stageKV(ktile + 1, (ktile + 1) & 1);

        const _Float16* Kb = Ks[ktile & 1];
        const _Float16* Vb = Vs[ktile & 1];

        // S^T[key][q]: 4 key tiles x (2 x 32-deep d chunks)
        v8f s[4];
#pragma unroll
        for (int mt = 0; mt < 4; ++mt) {
            s[mt] = (v8f){};
#pragma unroll
            for (int dt = 0; dt < 2; ++dt) {
                V16 a;
                const int arow = (mt * 16 + l16) * 64 + dt * 32;
                a.h[0] = *(const v8h*)&Kb[arow + hi * 8];
                a.h[1] = *(const v8h*)&Kb[arow + 16 + hi * 8];
                s[mt] = __builtin_amdgcn_wmma_f32_16x16x32_f16(
                    false, a.v, false, bq[dt].v, (short)0, s[mt], false, false);
            }
        }

        // online softmax; each lane tracks stats for q = l16 (32 keys here,
        // complementary 32 in the other half-wave -> one xor-16 combine).
        // Log-depth max tree to shorten the VALU dependency chain.
        float tmv[4];
#pragma unroll
        for (int mt = 0; mt < 4; ++mt) {
            float a0 = fmaxf(s[mt][0], s[mt][1]);
            float a1 = fmaxf(s[mt][2], s[mt][3]);
            float a2 = fmaxf(s[mt][4], s[mt][5]);
            float a3 = fmaxf(s[mt][6], s[mt][7]);
            tmv[mt] = fmaxf(fmaxf(a0, a1), fmaxf(a2, a3));
        }
        float tm = fmaxf(fmaxf(tmv[0], tmv[1]), fmaxf(tmv[2], tmv[3]));
        tm = fmaxf(tm, __shfl_xor(tm, 16));
        const float nmax  = fmaxf(rmax, tm);
        const float alpha = __expf(rmax - nmax);
        rmax = nmax;

        float tsv[4];
#pragma unroll
        for (int mt = 0; mt < 4; ++mt) {
#pragma unroll
            for (int r = 0; r < 8; ++r) s[mt][r] = __expf(s[mt][r] - nmax);
            float a0 = s[mt][0] + s[mt][1];
            float a1 = s[mt][2] + s[mt][3];
            float a2 = s[mt][4] + s[mt][5];
            float a3 = s[mt][6] + s[mt][7];
            tsv[mt] = (a0 + a1) + (a2 + a3);
        }
        float ts = (tsv[0] + tsv[1]) + (tsv[2] + tsv[3]);
        ts += __shfl_xor(ts, 16);
        rsum = rsum * alpha + ts;

        // rescale O: row r of O is q = r + 8*hi; fetch that row's alpha
#pragma unroll
        for (int r = 0; r < 8; ++r) {
            float alr = __shfl(alpha, r + hi * 8);
#pragma unroll
            for (int nt = 0; nt < 4; ++nt) o[nt][r] *= alr;
        }

        // P A-fragments straight from score fragments (no LDS round-trip):
        // chunk0 keys 0..31 = tiles {0,1}; chunk1 keys 32..63 = tiles {2,3}
        V16 ap[2];
#pragma unroll
        for (int e = 0; e < 8; ++e) {
            ap[0].v[e]     = (_Float16)s[0][e];
            ap[0].v[e + 8] = (_Float16)s[1][e];
            ap[1].v[e]     = (_Float16)s[2][e];
            ap[1].v[e + 8] = (_Float16)s[3][e];
        }

        // O[q][d] += P . V
#pragma unroll
        for (int dt = 0; dt < 2; ++dt) {
            const int kb = dt * 32 + hi * 8;
#pragma unroll
            for (int nt = 0; nt < 4; ++nt) {
                V16 bv;
#pragma unroll
                for (int e = 0; e < 8; ++e) {
                    bv.v[e]     = Vb[(kb + e) * 64 + nt * 16 + l16];
                    bv.v[e + 8] = Vb[(kb + 16 + e) * 64 + nt * 16 + l16];
                }
                o[nt] = __builtin_amdgcn_wmma_f32_16x16x32_f16(
                    false, ap[dt].v, false, bv.v, (short)0, o[nt], false, false);
            }
        }
    }

    // normalize + store: O row r is q = wave*16 + r + 8*hi, col d = nt*16+l16
    float inv[8];
#pragma unroll
    for (int r = 0; r < 8; ++r) inv[r] = 1.0f / __shfl(rsum, r + hi * 8);
#pragma unroll
    for (int r = 0; r < 8; ++r) {
        int token = qt * 64 + wave * 16 + hi * 8 + r;
        size_t base = headBase + (size_t)token * tokStride + l16;
        O[base +  0] = (_Float16)(o[0][r] * inv[r]);
        O[base + 16] = (_Float16)(o[1][r] * inv[r]);
        O[base + 32] = (_Float16)(o[2][r] * inv[r]);
        O[base + 48] = (_Float16)(o[3][r] * inv[r]);
    }
}

// ---------------------------------------------------------------------------
// launch
// ---------------------------------------------------------------------------
extern "C" void kernel_launch(void* const* d_in, const int* in_sizes, int n_in,
                              void* d_out, int out_size, void* d_ws, size_t ws_size,
                              hipStream_t stream) {
    const float* x  = (const float*)d_in[0];
    const float* wq = (const float*)d_in[1];
    const float* bq = (const float*)d_in[2];
    const float* wk = (const float*)d_in[3];
    const float* bk = (const float*)d_in[4];
    const float* wv = (const float*)d_in[5];
    const float* bv = (const float*)d_in[6];
    const float* wo = (const float*)d_in[7];
    const float* bo = (const float*)d_in[8];
    float* out = (float*)d_out;

    const int B = 4, L = 2048, D = 1024, H = 16;
    const int M = B * L;        // 8192 tokens
    const int W = D * D;        // 1024*1024 weight elements

    _Float16* ws = (_Float16*)d_ws;
    size_t off = 0;
    _Float16* xh  = ws + off; off += (size_t)M * D;
    _Float16* wqh = ws + off; off += (size_t)W;
    _Float16* wkh = ws + off; off += (size_t)W;
    _Float16* wvh = ws + off; off += (size_t)W;
    _Float16* woh = ws + off; off += (size_t)W;
    _Float16* qh  = ws + off; off += (size_t)M * D;
    _Float16* kh  = ws + off; off += (size_t)M * D;
    _Float16* vh  = ws + off; off += (size_t)M * D;
    _Float16* oh  = ws + off; off += (size_t)M * D;

    // f32 -> f16
    cvt_f32_to_f16_kernel<<<2048, 256, 0, stream>>>(x,  xh,  M * D);
    cvt_f32_to_f16_kernel<<<1024, 256, 0, stream>>>(wq, wqh, W);
    cvt_f32_to_f16_kernel<<<1024, 256, 0, stream>>>(wk, wkh, W);
    cvt_f32_to_f16_kernel<<<1024, 256, 0, stream>>>(wv, wvh, W);
    cvt_f32_to_f16_kernel<<<1024, 256, 0, stream>>>(wo, woh, W);

    // QKV projections (scale 1/sqrt(64) folded into Q)
    dim3 ggrid(M / 128, D / 128);
    gemm_bias_kernel<_Float16><<<ggrid, 256, 0, stream>>>(xh, wqh, bq, qh, M, D, D, 0.125f);
    gemm_bias_kernel<_Float16><<<ggrid, 256, 0, stream>>>(xh, wkh, bk, kh, M, D, D, 1.0f);
    gemm_bias_kernel<_Float16><<<ggrid, 256, 0, stream>>>(xh, wvh, bv, vh, M, D, D, 1.0f);

    // flash attention
    flash_attn_kernel<<<B * H * (L / 64), 128, 0, stream>>>(qh, kh, vh, oh, L, H);

    // output projection -> f32
    gemm_bias_kernel<float><<<ggrid, 256, 0, stream>>>(oh, woh, bo, out, M, D, D, 1.0f);
}
